// GeometricEmbedding_25623774888098
// MI455X (gfx1250) — matrix operand order repaired and checked by reading, not compiled, not run on hardware
//
#include <hip/hip_runtime.h>
#include <hip/hip_bf16.h>

typedef __attribute__((ext_vector_type(16))) __bf16 v16bf;
typedef __attribute__((ext_vector_type(8)))  float  v8f;
typedef __attribute__((ext_vector_type(4)))  unsigned int v4u;
typedef __attribute__((ext_vector_type(8)))  int v8i;
typedef __attribute__((ext_vector_type(4)))  int v4i;

constexpr int kB = 32, kN = 8192, kD = 256;
constexpr int TILE   = 128;            // columns per workgroup
constexpr int SLABC  = 32;             // channels per K-slab
constexpr int NSLAB  = kD / SLABC;     // 8
constexpr int NCHUNK = NSLAB * 3;      // 24 WMMA K-chunks (slab x tap)
constexpr int STRH   = 40;             // bf16 units per staged column (32 + 8 pad)
constexpr int RAWW   = 130;            // staged columns per slab (1-col halo each side)
constexpr int WLEN_ROT   = 4 * kD * 3; // 3072 floats
constexpr int WLEN_TRANS = 2 * kD * 3; // 1536 floats
constexpr float LRELU = 0.2f;

// LDS carving (bytes): Afrag | bf16 slab x2 | raw f32 slab ; wlds overlays bf
constexpr int AFRAG_B = NCHUNK * 32 * 16 * 2;          // 24576
constexpr int BFBUF_B = 132 * STRH * 2;                // 10560 (per buffer)
constexpr int RAW_B   = SLABC * RAWW * 4;              // 16640
constexpr int SMEM_B  = AFRAG_B + 2 * BFBUF_B + RAW_B; // 62336 (< 64KB)

#if defined(__HIP_DEVICE_COMPILE__) && \
    __has_builtin(__builtin_amdgcn_tensor_load_to_lds) && \
    __has_builtin(__builtin_amdgcn_s_wait_tensorcnt)
#define USE_TDM 1
#else
#define USE_TDM 0
#endif

__device__ __forceinline__ unsigned short f32_to_bf16_rne(float f) {
    unsigned int x = __float_as_uint(f);
    unsigned int r = (x + 0x7FFFu + ((x >> 16) & 1u)) >> 16;
    return (unsigned short)r;
}

#if USE_TDM
// 2-D TDM load of f32 tile (tile0 x rows) from global to LDS per CDNA5 ISA.
// OOB columns (x >= tensor_dim0) read as zero -> free right-edge zero padding.
__device__ __forceinline__ void tdm_load_2d_f32(unsigned lds_off_bytes,
                                                const void* gsrc,
                                                unsigned tensor_dim0,
                                                unsigned tile0, unsigned rows,
                                                unsigned stride0) {
    unsigned long long ga = (unsigned long long)(uintptr_t)gsrc;
    v4u g0 = { 1u,                                  // count=1
               lds_off_bytes,
               (unsigned)(ga & 0xffffffffu),
               (unsigned)(((ga >> 32) & 0x01ffffffu) | 0x80000000u) }; // type=2
    v8i g1 = { (int)0x00020000u,                           // data_size = 4B
               (int)((tensor_dim0 & 0xffffu) << 16),       // td0 lo16
               (int)((tensor_dim0 >> 16) | (rows << 16)),  // td0 hi16 | td1 lo16
               (int)(tile0 << 16),                         // td1 hi=0 | tile_dim0
               (int)rows,                                  // tile_dim1 | tile_dim2=0
               (int)stride0,                               // dim0 stride lo32
               0, 0 };
    v4i z4 = {0, 0, 0, 0};
#if defined(__clang_major__) && (__clang_major__ >= 23)
    v8i z8 = {0, 0, 0, 0, 0, 0, 0, 0};
    __builtin_amdgcn_tensor_load_to_lds(g0, g1, z4, z4, z8, 0);
#else
    __builtin_amdgcn_tensor_load_to_lds(g0, g1, z4, z4, 0);
#endif
}
#endif

// offset channels (18) from the 6 conv outputs: tm = y[0..3] (2x2), trans = y[4..5]
__device__ __forceinline__ void offset_from_y6(const float y[6], float off[18]) {
    const float tm00 = y[0], tm01 = y[1], tm10 = y[2], tm11 = y[3];
    const float t0 = y[4], t1 = y[5];
#pragma unroll
    for (int k = 0; k < 9; ++k) {
        float r0 = (float)(k / 3 - 1);   // REG row0: -1,-1,-1,0,0,0,1,1,1
        float r1 = (float)(k % 3 - 1);   // REG row1: -1,0,1,-1,0,1,-1,0,1
        off[2 * k]     = tm00 * r0 + tm01 * r1 - r0 + t0;
        off[2 * k + 1] = tm10 * r0 + tm11 * r1 - r1 + t1;
    }
}

// ---------------------------------------------------------------------------
// k1: rot(4ch)+trans(2ch) conv1d(K=3,pad=1) over D=256 as 16x768 * 768xN GEMM
// on v_wmma_f32_16x16x32_bf16.  TDM DMAs both the weights and every feats
// slab into LDS (double-buffered against compute); inner loop = ds_load+wmma.
// ---------------------------------------------------------------------------
__global__ void __launch_bounds__(256)
k1_conv_wmma(const float* __restrict__ feats,
             const float* __restrict__ rot_w,  const float* __restrict__ rot_b,
             const float* __restrict__ trans_w,const float* __restrict__ trans_b,
             float* __restrict__ y6)
{
    __shared__ __align__(32) unsigned char smem[SMEM_B];
    unsigned short* Afrag  = (unsigned short*)smem;                    // 24.0 KB
    unsigned short* bfslab = (unsigned short*)(smem + AFRAG_B);        // 2x10.3 KB
    float*          raw    = (float*)(smem + AFRAG_B + 2 * BFBUF_B);   // 16.25 KB
    float*          wlds   = (float*)(smem + AFRAG_B);                 // 18 KB overlay

    const int tid  = threadIdx.x;
    const int lane = tid & 31;
    const int wv   = tid >> 5;                 // 0..7 (wave32)
    const int b    = blockIdx.x / (kN / TILE);
    const int tile = blockIdx.x % (kN / TILE);
    const int n0   = tile * TILE;
    const bool leftEdge = (tile == 0);         // cannot express n = -1 via TDM

    const float* fb = feats + (size_t)b * kD * kN;

    // ---- stage weights into LDS (TDM where available) ----
#if USE_TDM
    if (wv == 0) {
        unsigned wbase = (unsigned)(unsigned long long)(uintptr_t)&wlds[0];
        tdm_load_2d_f32(wbase, rot_w, WLEN_ROT, WLEN_ROT, 1, WLEN_ROT);
        tdm_load_2d_f32(wbase + WLEN_ROT * 4, trans_w, WLEN_TRANS, WLEN_TRANS, 1,
                        WLEN_TRANS);
        __builtin_amdgcn_s_wait_tensorcnt(0);
    }
#else
    for (int idx = tid; idx < WLEN_ROT; idx += 256)   wlds[idx] = rot_w[idx];
    for (int idx = tid; idx < WLEN_TRANS; idx += 256) wlds[WLEN_ROT + idx] = trans_w[idx];
#endif
    __syncthreads();

    // ---- build bf16 A-fragments from LDS weights (once per block) ----
    for (int idx = tid; idx < NCHUNK * 32 * 16; idx += 256) {
        int e     = idx & 15;
        int ln    = (idx >> 4) & 31;
        int chunk = idx >> 9;
        int s  = chunk / 3;
        int kk = chunk - s * 3;
        int row = ln & 15;                               // out channel, 6 used
        // A K order used here (contiguous, same as B): K = e + (lane>=16)*16
        int kloc = e + ((ln >> 4) << 4);
        int c = s * SLABC + kloc;                        // input channel 0..255
        float wval = 0.0f;
        if (row < 4)       wval = wlds[row * (kD * 3) + c * 3 + kk];
        else if (row < 6)  wval = wlds[WLEN_ROT + (row - 4) * (kD * 3) + c * 3 + kk];
        Afrag[idx] = f32_to_bf16_rne(wval);
    }
    __syncthreads();   // Afrag done; wlds overlay dead, bf/raw regions free

    // fetch slab s into raw[c][j] (pitch RAWW), j=0 is column n0-1
    auto fetch = [&](int s) {
        if (leftEdge) {   // per-thread fallback: column -1 zero-padded
            for (int idx = tid; idx < SLABC * RAWW; idx += 256) {
                int c = idx / RAWW, j = idx - c * RAWW;
                raw[idx] = (j >= 1) ? fb[(size_t)(s * SLABC + c) * kN + (j - 1)] : 0.0f;
            }
        } else {
#if USE_TDM
            if (wv == 0) {
                unsigned roff = (unsigned)(unsigned long long)(uintptr_t)&raw[0];
                const float* src = fb + (size_t)(s * SLABC) * kN + (n0 - 1);
                tdm_load_2d_f32(roff, src, (unsigned)(kN - (n0 - 1)),
                                RAWW, SLABC, kN);
            }
#else
            for (int idx = tid; idx < SLABC * RAWW; idx += 256) {
                int c = idx / RAWW, j = idx - c * RAWW;
                int n = n0 - 1 + j;
                raw[idx] = (n < kN) ? fb[(size_t)(s * SLABC + c) * kN + n] : 0.0f;
            }
#endif
        }
    };

    const int colL = wv * 16 + (lane & 15);              // 0..127
    const int hiK  = lane >> 4;                          // B: K = hiK*16 + e

    v8f acc = {0.f, 0.f, 0.f, 0.f, 0.f, 0.f, 0.f, 0.f};

    fetch(0);
    for (int s = 0; s < NSLAB; ++s) {
#if USE_TDM
        if (wv == 0 && !leftEdge) __builtin_amdgcn_s_wait_tensorcnt(0);
#endif
        __syncthreads();                                 // raw[s] visible to all

        // convert + transpose raw f32 -> bf16 slab (double-buffered)
        unsigned short* bfs = bfslab + (s & 1) * (132 * STRH);
        for (int idx = tid; idx < SLABC * RAWW; idx += 256) {
            int c = idx / RAWW, j = idx - c * RAWW;
            bfs[j * STRH + c] = f32_to_bf16_rne(raw[idx]);
        }
        __syncthreads();                                 // bf ready, raw free

        if (s + 1 < NSLAB) fetch(s + 1);                 // TDM overlaps compute

#pragma unroll
        for (int kk = 0; kk < 3; ++kk) {
            int chunk = s * 3 + kk;
            v16bf afrag = *reinterpret_cast<const v16bf*>(&Afrag[(chunk * 32 + lane) * 16]);
            const unsigned short* bp = &bfs[(colL + kk) * STRH + hiK * 16];
            union { v16bf v; uint4 q[2]; } bf;
            bf.q[0] = *reinterpret_cast<const uint4*>(bp);
            bf.q[1] = *reinterpret_cast<const uint4*>(bp + 8);
            acc = __builtin_amdgcn_wmma_f32_16x16x32_bf16(
                false, afrag, false, bf.v, (short)0, acc, false, false);
        }
    }

    // C/D layout: lanes 0..15 hold rows 0..7 in acc[0..7]; rows 0..5 live
    if (lane < 16) {
        int n = n0 + colL;
#pragma unroll
        for (int r = 0; r < 6; ++r) {
            float bias = (r < 4) ? rot_b[r] : trans_b[r - 4];
            y6[((size_t)b * 6 + r) * kN + n] = acc[r] + bias;
        }
    }
}

// ---------------------------------------------------------------------------
__global__ void k0_zero(float* stats) {
    if (threadIdx.x < 8) stats[threadIdx.x] = 0.0f;
}

// ---------------------------------------------------------------------------
// k2: pos conv (20ch -> 2ch, K=3) + per-channel sum / sumsq for BN1
// ---------------------------------------------------------------------------
__global__ void __launch_bounds__(256)
k2_posconv_stats(const float* __restrict__ pts, const float* __restrict__ y6,
                 const float* __restrict__ pos_w, const float* __restrict__ pos_b,
                 float* __restrict__ zbuf, float* __restrict__ stats)
{
    __shared__ float sred[4][256];
    const int g = blockIdx.x * 256 + threadIdx.x;      // exact B*N grid
    const int b = g / kN, n = g % kN;

    float z0 = pos_b[0], z1 = pos_b[1];
#pragma unroll
    for (int kk = 0; kk < 3; ++kk) {
        int p = n + kk - 1;
        if (p < 0 || p >= kN) continue;                // zero pad
        float x20[20];
        x20[0] = pts[((size_t)b * 2 + 0) * kN + p];
        x20[1] = pts[((size_t)b * 2 + 1) * kN + p];
        float y[6];
#pragma unroll
        for (int r = 0; r < 6; ++r) y[r] = y6[((size_t)b * 6 + r) * kN + p];
        offset_from_y6(y, &x20[2]);
#pragma unroll
        for (int cc = 0; cc < 20; ++cc) {
            z0 += x20[cc] * pos_w[0 * 60 + cc * 3 + kk];
            z1 += x20[cc] * pos_w[1 * 60 + cc * 3 + kk];
        }
    }
    zbuf[((size_t)b * 2 + 0) * kN + n] = z0;
    zbuf[((size_t)b * 2 + 1) * kN + n] = z1;

    sred[0][threadIdx.x] = z0;      sred[1][threadIdx.x] = z1;
    sred[2][threadIdx.x] = z0 * z0; sred[3][threadIdx.x] = z1 * z1;
    __syncthreads();
    for (int st = 128; st > 0; st >>= 1) {
        if (threadIdx.x < st)
#pragma unroll
            for (int q = 0; q < 4; ++q)
                sred[q][threadIdx.x] += sred[q][threadIdx.x + st];
        __syncthreads();
    }
    if (threadIdx.x == 0)
#pragma unroll
        for (int q = 0; q < 4; ++q) atomicAdd(&stats[q], sred[q][0]);
}

// ---------------------------------------------------------------------------
// k3: BN1 + leaky + per-point rotation + ori conv (2ch, K=3) + BN2 stats
// ---------------------------------------------------------------------------
__global__ void __launch_bounds__(256)
k3_rot_oriconv(const float* __restrict__ y6, const float* __restrict__ zbuf,
               const float* __restrict__ pos_g, const float* __restrict__ pos_beta,
               const float* __restrict__ ori_w, const float* __restrict__ ori_b,
               const float* __restrict__ stats,
               float* __restrict__ wbuf, float* __restrict__ stats2)
{
    __shared__ float sred[4][256];
    const float cnt = (float)(kB * kN);
    const float m0 = stats[0] / cnt, m1 = stats[1] / cnt;
    const float v0 = stats[2] / cnt - m0 * m0;
    const float v1 = stats[3] / cnt - m1 * m1;
    const float g0 = pos_g[0] * rsqrtf(v0 + 1e-5f);
    const float g1 = pos_g[1] * rsqrtf(v1 + 1e-5f);
    const float be0 = pos_beta[0], be1 = pos_beta[1];

    const int g = blockIdx.x * 256 + threadIdx.x;
    const int b = g / kN, n = g % kN;

    float w0 = ori_b[0], w1 = ori_b[1];
#pragma unroll
    for (int kk = 0; kk < 3; ++kk) {
        int p = n + kk - 1;
        if (p < 0 || p >= kN) continue;
        float z0 = zbuf[((size_t)b * 2 + 0) * kN + p];
        float z1 = zbuf[((size_t)b * 2 + 1) * kN + p];
        float x0 = g0 * (z0 - m0) + be0; x0 = (x0 >= 0.f) ? x0 : LRELU * x0;
        float x1 = g1 * (z1 - m1) + be1; x1 = (x1 >= 0.f) ? x1 : LRELU * x1;

        float y[6];
#pragma unroll
        for (int r = 0; r < 6; ++r) y[r] = y6[((size_t)b * 6 + r) * kN + p];
        float off[18]; offset_from_y6(y, off);
        float mr = -1e30f;
#pragma unroll
        for (int k = 0; k < 9; ++k) {
            float rad = atanf((off[2 * k + 1] + 1e-6f) / (off[2 * k] + 1e-6f));
            mr = fmaxf(mr, rad);
        }
        float cc_ = cosf(mr), ss_ = sinf(mr);
        float u0 = x0 * cc_ - x1 * ss_;    // out[0] = x0*c + x1*(-s)
        float u1 = x0 * ss_ + x1 * cc_;    // out[1] = x0*s + x1*c
        w0 += u0 * ori_w[0 * 6 + 0 * 3 + kk] + u1 * ori_w[0 * 6 + 1 * 3 + kk];
        w1 += u0 * ori_w[1 * 6 + 0 * 3 + kk] + u1 * ori_w[1 * 6 + 1 * 3 + kk];
    }
    wbuf[((size_t)b * 2 + 0) * kN + n] = w0;
    wbuf[((size_t)b * 2 + 1) * kN + n] = w1;

    sred[0][threadIdx.x] = w0;      sred[1][threadIdx.x] = w1;
    sred[2][threadIdx.x] = w0 * w0; sred[3][threadIdx.x] = w1 * w1;
    __syncthreads();
    for (int st = 128; st > 0; st >>= 1) {
        if (threadIdx.x < st)
#pragma unroll
            for (int q = 0; q < 4; ++q)
                sred[q][threadIdx.x] += sred[q][threadIdx.x + st];
        __syncthreads();
    }
    if (threadIdx.x == 0)
#pragma unroll
        for (int q = 0; q < 4; ++q) atomicAdd(&stats2[q], sred[q][0]);
}

// ---------------------------------------------------------------------------
// k4: final batchnorm -> out (B,2,N)
// ---------------------------------------------------------------------------
__global__ void __launch_bounds__(256)
k4_final(const float* __restrict__ wbuf,
         const float* __restrict__ ori_g, const float* __restrict__ ori_beta,
         const float* __restrict__ stats2, float* __restrict__ out)
{
    const float cnt = (float)(kB * kN);
    const float m0 = stats2[0] / cnt, m1 = stats2[1] / cnt;
    const float v0 = stats2[2] / cnt - m0 * m0;
    const float v1 = stats2[3] / cnt - m1 * m1;
    const float g0 = ori_g[0] * rsqrtf(v0 + 1e-5f);
    const float g1 = ori_g[1] * rsqrtf(v1 + 1e-5f);

    const int g = blockIdx.x * 256 + threadIdx.x;
    const int b = g / kN, n = g % kN;
    float w0 = wbuf[((size_t)b * 2 + 0) * kN + n];
    float w1 = wbuf[((size_t)b * 2 + 1) * kN + n];
    out[((size_t)b * 2 + 0) * kN + n] = g0 * (w0 - m0) + ori_beta[0];
    out[((size_t)b * 2 + 1) * kN + n] = g1 * (w1 - m1) + ori_beta[1];
}

extern "C" void kernel_launch(void* const* d_in, const int* in_sizes, int n_in,
                              void* d_out, int out_size, void* d_ws, size_t ws_size,
                              hipStream_t stream)
{
    (void)in_sizes; (void)n_in; (void)out_size; (void)ws_size;
    const float* pts      = (const float*)d_in[0];
    const float* feats    = (const float*)d_in[1];
    const float* rot_w    = (const float*)d_in[2];
    const float* rot_b    = (const float*)d_in[3];
    const float* trans_w  = (const float*)d_in[4];
    const float* trans_b  = (const float*)d_in[5];
    const float* pos_w    = (const float*)d_in[6];
    const float* pos_b    = (const float*)d_in[7];
    const float* pos_g    = (const float*)d_in[8];
    const float* pos_beta = (const float*)d_in[9];
    const float* ori_w    = (const float*)d_in[10];
    const float* ori_b    = (const float*)d_in[11];
    const float* ori_g    = (const float*)d_in[12];
    const float* ori_beta = (const float*)d_in[13];

    float* ws    = (float*)d_ws;
    float* y6    = ws;                                   // 32*6*8192
    float* zbuf  = y6   + (size_t)kB * 6 * kN;           // 32*2*8192
    float* wbuf  = zbuf + (size_t)kB * 2 * kN;           // 32*2*8192
    float* stats = wbuf + (size_t)kB * 2 * kN;           // 8 floats
    float* out   = (float*)d_out;

    const int npt_blocks = (kB * kN) / 256;              // 1024

    k0_zero<<<1, 32, 0, stream>>>(stats);
    k1_conv_wmma<<<kB * (kN / TILE), 256, 0, stream>>>(feats, rot_w, rot_b,
                                                       trans_w, trans_b, y6);
    k2_posconv_stats<<<npt_blocks, 256, 0, stream>>>(pts, y6, pos_w, pos_b,
                                                     zbuf, stats);
    k3_rot_oriconv<<<npt_blocks, 256, 0, stream>>>(y6, zbuf, pos_g, pos_beta,
                                                   ori_w, ori_b, stats,
                                                   wbuf, stats + 4);
    k4_final<<<npt_blocks, 256, 0, stream>>>(wbuf, ori_g, ori_beta, stats + 4, out);
}
